// MultiAttnMatch_7052336300536
// MI455X (gfx1250) — matrix operand order repaired and checked by reading, not compiled, not run on hardware
//
#include <hip/hip_runtime.h>
#include <hip/hip_bf16.h>
#include <cstdint>

// Problem constants (match reference)
#define BB   16
#define L1N  1024
#define L2N  1024
#define DINN 1024
#define HN   16
#define DKN  64
#define DVN  64

typedef __attribute__((ext_vector_type(16))) __bf16       v16bf;
typedef __attribute__((ext_vector_type(8)))  float        v8f;
typedef __attribute__((ext_vector_type(4)))  unsigned int ui32x4;
typedef __attribute__((ext_vector_type(8)))  int          i32x8;
typedef __attribute__((ext_vector_type(4)))  int          i32x4;

union Frag {
    v16bf v;
    unsigned int u[8];
    uint4 q[2];
};

#if defined(__has_builtin)
#if __has_builtin(__builtin_amdgcn_tensor_load_to_lds) && \
    __has_builtin(__builtin_amdgcn_s_wait_tensorcnt)
#define HAVE_TDM 1
#endif
#endif
#ifndef HAVE_TDM
#define HAVE_TDM 0
#endif

#if HAVE_TDM
// ---------------------------------------------------------------------------
// 2D TDM load: tile (tile_x elems wide, tile_y rows) of bf16 from a row-major
// tensor with row stride stride_x (elems) into LDS at lds_off, with optional
// LDS row padding (ISA D# groups, CDNA5 ISA).
// This toolchain exposes the 6-arg builtin:
//   (uint32x4 g0, int32x8 g1, int32x4 g2, int32x4 g3, int32x8, i32 cpol)
// ---------------------------------------------------------------------------
__device__ __forceinline__ void tdm_load_2d(unsigned lds_off, const void* gptr,
                                            unsigned tile_x, unsigned tile_y,
                                            unsigned stride_x,
                                            unsigned pad_interval,
                                            unsigned pad_amount, bool pad_en)
{
    unsigned long long ga = (unsigned long long)(uintptr_t)gptr;
    union { unsigned int a[4]; ui32x4 v; } g0 = {};
    union { int a[8];          i32x8  v; } g1 = {};
    union { int a[4];          i32x4  v; } gz4 = {};
    union { int a[8];          i32x8  v; } gz8 = {};

    // Group 0: count=1 (valid), lds_addr, global_addr[56:0], type=2 (image)
    g0.a[0] = 1u;
    g0.a[1] = lds_off;
    g0.a[2] = (unsigned)(ga & 0xffffffffu);
    g0.a[3] = (unsigned)((ga >> 32) & 0x1ffffffu) | (2u << 30);

    const unsigned tdim0 = stride_x;      // bounds >= tile -> no OOB zeroing
    const unsigned tdim1 = 1u << 20;
    // Group 1: data_size=1 (2B), pad ctl, tensor/tile dims, dim0 stride
    g1.a[0] = (int)((1u << 16) | (pad_en ? (1u << 20) : 0u) |
                    (pad_interval << 22) | (pad_amount << 25));
    g1.a[1] = (int)((tdim0 & 0xffffu) << 16);
    g1.a[2] = (int)((tdim0 >> 16) | ((tdim1 & 0xffffu) << 16));
    g1.a[3] = (int)((tdim1 >> 16) | ((tile_x & 0xffffu) << 16));
    g1.a[4] = (int)(tile_y & 0xffffu);
    g1.a[5] = (int)stride_x;              // tensor_dim0_stride[31:0]
    g1.a[6] = 0;
    g1.a[7] = 0;

    __builtin_amdgcn_tensor_load_to_lds(g0.v, g1.v, gz4.v, gz4.v, gz8.v, 0);
}
#endif

// ---------------------------------------------------------------------------
// Kernel 0: fp32 -> bf16 cast of x (grid.y==0) and y (grid.y==1), so the GEMM
// A-operands are directly TDM-transferable (TDM cannot convert dtypes).
// ---------------------------------------------------------------------------
__global__ __launch_bounds__(256)
void cast_kernel(const float* __restrict__ xa, const float* __restrict__ yb,
                 __bf16* __restrict__ xo, __bf16* __restrict__ yo)
{
    const float* src = blockIdx.y ? yb : xa;
    __bf16*      dst = blockIdx.y ? yo : xo;
    size_t i = ((size_t)blockIdx.x * 256 + threadIdx.x) * 8;
    float4 f0 = *(const float4*)(src + i);
    float4 f1 = *(const float4*)(src + i + 4);
    union { __bf16 h[8]; uint4 q; } pk;
    pk.h[0] = (__bf16)f0.x; pk.h[1] = (__bf16)f0.y;
    pk.h[2] = (__bf16)f0.z; pk.h[3] = (__bf16)f0.w;
    pk.h[4] = (__bf16)f1.x; pk.h[5] = (__bf16)f1.y;
    pk.h[6] = (__bf16)f1.z; pk.h[7] = (__bf16)f1.w;
    *(uint4*)(dst + i) = pk.q;
}

// ---------------------------------------------------------------------------
// Kernel 1: fused QKV projection.  grid.z: 0=Q(from x), 1=K(from y), 2=V(from y)
// Block = 256 threads (8 waves). Tile: 128(M) x 64(N), K-step 32.
// A tile arrives via TDM (padded to 34-elem rows); B tile is hand-transposed.
// Outputs bf16: Q [B,H,L1,DK] (pre-scaled 1/8), K [B,H,L2,DK], V^T [B,H,DV,L2].
// ---------------------------------------------------------------------------
__global__ __launch_bounds__(256)
void qkv_proj_kernel(const __bf16* __restrict__ xbf,
                     const __bf16* __restrict__ ybf,
                     const float* __restrict__ Wq, const float* __restrict__ bq,
                     const float* __restrict__ Wk, const float* __restrict__ bk,
                     const float* __restrict__ Wv, const float* __restrict__ bv,
                     __bf16* __restrict__ qo, __bf16* __restrict__ ko,
                     __bf16* __restrict__ vo)
{
    const int which = blockIdx.z;
    const __bf16* in  = (which == 0) ? xbf : ybf;
    const float* W    = (which == 0) ? Wq : (which == 1) ? Wk : Wv;
    const float* bias = (which == 0) ? bq : (which == 1) ? bk : bv;
    const float scale = (which == 0) ? 0.125f : 1.0f;  // 1/sqrt(64) into Q

    const int mBase = blockIdx.x * 128;
    const int nBase = blockIdx.y * 64;

    __shared__ __bf16 aT[128 * 34];   // 34-elem rows (TDM pad: 16dw + 1dw)
    __shared__ __bf16 bT[64 * 36];    // B stored [n][k], padded

    const int tid  = threadIdx.x;
    const int lane = tid & 31;
    const int wave = tid >> 5;
    const int nidx = lane & 15;
    const int half = lane >> 4;

    v8f acc[4] = {};

    for (int k0 = 0; k0 < DINN; k0 += 32) {
        const __bf16* asrc = in + (size_t)mBase * DINN + k0;
#if HAVE_TDM
        if (wave == 0) {   // async: 128x32 bf16 tile -> LDS, rows padded to 34
            tdm_load_2d((unsigned)(uintptr_t)aT, asrc, 32, 128, DINN,
                        /*pad_interval 16dw*/ 3, /*pad_amount 1dw*/ 0, true);
        }
#else
        #pragma unroll
        for (int p = 0; p < 16; ++p) {
            int rl = p * 8 + (tid >> 5);
            int kk = tid & 31;
            aT[rl * 34 + kk] = asrc[(size_t)rl * DINN + kk];
        }
#endif
        // B tile fill overlaps the in-flight TDM: fp32 W -> bf16, transposed
        #pragma unroll
        for (int p = 0; p < 8; ++p) {
            int kk = p * 4 + (tid >> 6);
            int nn = tid & 63;
            bT[nn * 36 + kk] =
                (__bf16)W[(size_t)(k0 + kk) * (HN * DKN) + nBase + nn];
        }
#if HAVE_TDM
        if (wave == 0) __builtin_amdgcn_s_wait_tensorcnt(0);
#endif
        __syncthreads();

        // A fragment 16x32: VGPR v -> K=16h+2v
        Frag a;
        const unsigned int* arow =
            (const unsigned int*)(aT + (wave * 16 + nidx) * 34);
        #pragma unroll
        for (int v = 0; v < 4; ++v) a.u[v]     = arow[8 * half + v];
        #pragma unroll
        for (int v = 0; v < 4; ++v) a.u[4 + v] = arow[8 * half + 4 + v];

        #pragma unroll
        for (int t = 0; t < 4; ++t) {   // B fragment: VGPR v -> K = 16h + 2v
            Frag bf;
            const unsigned int* brow =
                (const unsigned int*)(bT + (t * 16 + nidx) * 36);
            #pragma unroll
            for (int v = 0; v < 8; ++v) bf.u[v] = brow[8 * half + v];
            acc[t] = __builtin_amdgcn_wmma_f32_16x16x32_bf16(
                false, a.v, false, bf.v, (short)0, acc[t], false, false);
        }
        __syncthreads();   // all reads done before next TDM overwrite
    }

    // Epilogue: bias, scale, bf16 store
    const int mTile = mBase + wave * 16;
    #pragma unroll
    for (int t = 0; t < 4; ++t) {
        int nG = nBase + t * 16 + nidx;
        int hh = nG >> 6;
        int dk = nG & 63;
        float bb_ = bias[nG];
        if (which < 2) {
            __bf16* outp = (which == 0) ? qo : ko;
            #pragma unroll
            for (int r = 0; r < 8; ++r) {
                int gm = mTile + 8 * half + r;
                int bi = gm >> 10;
                int ll = gm & 1023;
                outp[(((size_t)bi * HN + hh) * L1N + ll) * DKN + dk] =
                    (__bf16)((acc[t][r] + bb_) * scale);
            }
        } else {            // V transposed: 8 consecutive l2 -> one b128 store
            int gm0 = mTile + 8 * half;
            int bi  = gm0 >> 10;
            int ll0 = gm0 & 1023;
            union { __bf16 h[8]; uint4 q; } pk;
            #pragma unroll
            for (int r = 0; r < 8; ++r) pk.h[r] = (__bf16)(acc[t][r] + bb_);
            *(uint4*)(vo + (((size_t)bi * HN + hh) * DVN + dk) * L2N + ll0) =
                pk.q;
        }
    }
}

// ---------------------------------------------------------------------------
// Kernel 2: flash attention, one wave per (b, h, 16-row L1 tile).
// K/V tiles stream through LDS via double-buffered TDM: issue chunk c+1,
// then s_wait_tensorcnt(2) -> in-order TENSORcnt guarantees chunk c landed.
// ---------------------------------------------------------------------------
__global__ __launch_bounds__(32)
void attn_kernel(const __bf16* __restrict__ q, const __bf16* __restrict__ k,
                 const __bf16* __restrict__ vt,
                 const unsigned char* __restrict__ msk,
                 float* __restrict__ out)
{
    const int lane   = threadIdx.x;
    const int n      = lane & 15;
    const int half   = lane >> 4;
    const int l1Base = blockIdx.x * 16;
    const int h      = blockIdx.y;
    const int b      = blockIdx.z;
    const size_t bh  = (size_t)b * HN + h;

#if HAVE_TDM
    __shared__ __bf16 kbuf[2][32 * 66];   // K chunk: 32 rows x 64, pad -> 66
    __shared__ __bf16 vbuf[2][64 * 34];   // V chunk: 64 rows x 32, pad -> 34
#endif
    __shared__ __bf16 p_lds[16 * 40];     // P staging (C-layout -> A-layout)

    // Q A-fragments for DK=64 (two K=32 slabs), contiguous 16B runs per lane
    Frag qa[2];
    const __bf16* qrow = q + (bh * L1N + l1Base + n) * DKN;
    #pragma unroll
    for (int kc = 0; kc < 2; ++kc) {
        qa[kc].q[0] = *(const uint4*)(qrow + kc * 32 + 16 * half);
        qa[kc].q[1] = *(const uint4*)(qrow + kc * 32 + 16 * half + 8);
    }

    float rmax[8], rsum[8];
    #pragma unroll
    for (int r = 0; r < 8; ++r) { rmax[r] = -1e30f; rsum[r] = 0.0f; }
    v8f o[4] = {};

    const __bf16* kbase = k  + bh * L2N * DKN;
    const __bf16* vbase = vt + bh * DVN * L2N;
    const unsigned char* mbase = msk + (size_t)b * L2N;  // jnp bool = 1 byte

#if HAVE_TDM
    // Prologue: async-fetch chunk 0
    tdm_load_2d((unsigned)(uintptr_t)kbuf[0], kbase, 64, 32, DKN, 4, 0, true);
    tdm_load_2d((unsigned)(uintptr_t)vbuf[0], vbase, 32, 64, L2N, 3, 0, true);
#endif

    for (int c = 0; c < L2N / 32; ++c) {
        const int c2 = c * 32;
#if HAVE_TDM
        const int cur = c & 1;
        if (c + 1 < L2N / 32) {    // issue next chunk, keep only it pending
            tdm_load_2d((unsigned)(uintptr_t)kbuf[cur ^ 1],
                        kbase + (size_t)(c2 + 32) * DKN, 64, 32, DKN, 4, 0, true);
            tdm_load_2d((unsigned)(uintptr_t)vbuf[cur ^ 1],
                        vbase + (c2 + 32), 32, 64, L2N, 3, 0, true);
            __builtin_amdgcn_s_wait_tensorcnt(2);
        } else {
            __builtin_amdgcn_s_wait_tensorcnt(0);
        }
        const __bf16* kb = kbuf[cur];
        const __bf16* vb = vbuf[cur];
#endif
        // S = Q @ K^T for 32 L2 columns (2 tiles x 2 K-slabs)
        v8f s[2] = {};
        #pragma unroll
        for (int t = 0; t < 2; ++t) {
            #pragma unroll
            for (int kc = 0; kc < 2; ++kc) {
                Frag kf;   // B-frag: VGPR v -> K = kc*32 + 16h + 2v
#if HAVE_TDM
                const unsigned int* krow =
                    (const unsigned int*)(kb + (t * 16 + n) * 66);
                #pragma unroll
                for (int v = 0; v < 8; ++v)
                    kf.u[v] = krow[kc * 16 + 8 * half + v];
#else
                const __bf16* krow =
                    kbase + (size_t)(c2 + t * 16 + n) * DKN + kc * 32 + 16 * half;
                kf.q[0] = *(const uint4*)(krow);
                kf.q[1] = *(const uint4*)(krow + 8);
#endif
                s[t] = __builtin_amdgcn_wmma_f32_16x16x32_bf16(
                    false, qa[kc].v, false, kf.v, (short)0, s[t], false, false);
            }
            if (mbase[c2 + t * 16 + n]) {   // column mask (lane owns column n)
                #pragma unroll
                for (int r = 0; r < 8; ++r) s[t][r] = -1e30f;
            }
        }

        // Online softmax: xor-butterfly row reductions (bits 0..3 stay inside
        // each 16-lane half, matching C-layout row striping)
        float cmax[8], csum[8];
        #pragma unroll
        for (int r = 0; r < 8; ++r) cmax[r] = fmaxf(s[0][r], s[1][r]);
        #pragma unroll
        for (int d = 1; d < 16; d <<= 1) {
            #pragma unroll
            for (int r = 0; r < 8; ++r)
                cmax[r] = fmaxf(cmax[r], __shfl_xor(cmax[r], d, 32));
        }
        #pragma unroll
        for (int r = 0; r < 8; ++r) {
            float nm    = fmaxf(rmax[r], cmax[r]);
            float alpha = __expf(rmax[r] - nm);
            rmax[r] = nm;
            float p0 = __expf(s[0][r] - nm);
            float p1 = __expf(s[1][r] - nm);
            s[0][r] = p0; s[1][r] = p1;
            csum[r] = p0 + p1;
            #pragma unroll
            for (int t = 0; t < 4; ++t) o[t][r] *= alpha;
            rsum[r] *= alpha;
        }
        #pragma unroll
        for (int d = 1; d < 16; d <<= 1) {
            #pragma unroll
            for (int r = 0; r < 8; ++r)
                csum[r] += __shfl_xor(csum[r], d, 32);
        }
        #pragma unroll
        for (int r = 0; r < 8; ++r) rsum[r] += csum[r];

        // C-layout -> A-layout transpose of P through LDS (wave-private)
        #pragma unroll
        for (int r = 0; r < 8; ++r) {
            p_lds[(8 * half + r) * 40 + n]      = (__bf16)s[0][r];
            p_lds[(8 * half + r) * 40 + 16 + n] = (__bf16)s[1][r];
        }
        __syncthreads();
        Frag pf;
        const unsigned int* prow = (const unsigned int*)(p_lds + n * 40);
        #pragma unroll
        for (int v = 0; v < 4; ++v) pf.u[v]     = prow[8 * half + v];
        #pragma unroll
        for (int v = 0; v < 4; ++v) pf.u[4 + v] = prow[8 * half + 4 + v];
        __syncthreads();

        // O += P @ V
        #pragma unroll
        for (int t = 0; t < 4; ++t) {
            Frag vf;
#if HAVE_TDM
            const unsigned int* vrow =
                (const unsigned int*)(vb + (t * 16 + n) * 34);
            #pragma unroll
            for (int v = 0; v < 8; ++v) vf.u[v] = vrow[8 * half + v];
#else
            const __bf16* vrow =
                vbase + (size_t)(t * 16 + n) * L2N + c2 + 16 * half;
            vf.q[0] = *(const uint4*)(vrow);
            vf.q[1] = *(const uint4*)(vrow + 8);
#endif
            o[t] = __builtin_amdgcn_wmma_f32_16x16x32_bf16(
                false, pf.v, false, vf.v, (short)0, o[t], false, false);
        }
    }

    // Normalize and store fp32 output [B, L1, H*DV]
    #pragma unroll
    for (int t = 0; t < 4; ++t) {
        int dv = t * 16 + n;
        #pragma unroll
        for (int r = 0; r < 8; ++r) {
            int l1 = l1Base + 8 * half + r;
            out[(((size_t)b * L1N + l1) * HN + h) * DVN + dv] =
                o[t][r] / rsum[r];
        }
    }
}

// ---------------------------------------------------------------------------
extern "C" void kernel_launch(void* const* d_in, const int* in_sizes, int n_in,
                              void* d_out, int out_size, void* d_ws,
                              size_t ws_size, hipStream_t stream)
{
    (void)in_sizes; (void)n_in; (void)out_size; (void)ws_size;

    const float* x  = (const float*)d_in[0];
    const float* y  = (const float*)d_in[1];
    const unsigned char* ym = (const unsigned char*)d_in[2]; // jnp bool = i8
    const float* Wq = (const float*)d_in[3];
    const float* bq = (const float*)d_in[4];
    const float* Wk = (const float*)d_in[5];
    const float* bk = (const float*)d_in[6];
    const float* Wv = (const float*)d_in[7];
    const float* bv = (const float*)d_in[8];
    float* out = (float*)d_out;

    // Workspace: xbf, ybf (casts) + Q, K, V^T (bf16)  ~= 168 MB
    const size_t nTok = (size_t)BB * L1N * DINN;
    const size_t nQ   = (size_t)BB * HN * L1N * DKN;
    __bf16* xbf = (__bf16*)d_ws;
    __bf16* ybf = xbf + nTok;
    __bf16* qws = ybf + nTok;
    __bf16* kws = qws + nQ;
    __bf16* vws = kws + nQ;

    dim3 cg((unsigned)(nTok / (256 * 8)), 2);
    cast_kernel<<<cg, 256, 0, stream>>>(x, y, xbf, ybf);

    dim3 pg((BB * L1N) / 128, (HN * DKN) / 64, 3);
    qkv_proj_kernel<<<pg, 256, 0, stream>>>(xbf, ybf, Wq, bq, Wk, bk, Wv, bv,
                                            qws, kws, vws);

    dim3 ag(L1N / 16, HN, BB);
    attn_kernel<<<ag, 32, 0, stream>>>(qws, kws, vws, ym, out);
}
